// MegNetLayer_89043261981129
// MI455X (gfx1250) — matrix operand order repaired and checked by reading, not compiled, not run on hardware
//
#include <hip/hip_runtime.h>
#include <hip/hip_bf16.h>

// ---------------------------------------------------------------------------
// MegNet layer for MI455X (gfx1250): edge MLP (96->128->64->32) + scatter-mean
// + node MLP (64->128->64->32), all GEMMs on v_wmma_f32_16x16x32_f16.
// v2: A-fragments hoisted out of the column-tile loop (fewer ds loads),
//     float4-vectorized global fill/drain, packed f16 LDS stores in fill.
// ---------------------------------------------------------------------------

typedef _Float16 v16h __attribute__((ext_vector_type(16)));
typedef _Float16 v8h  __attribute__((ext_vector_type(8)));
typedef float    v8f  __attribute__((ext_vector_type(8)));

union HFrag { v16h v; v8h h[2]; };
union HPack4 { _Float16 h[4]; unsigned long long u; };

// One GEMM layer over a [128 x K_DIM] f16 activation tile in LDS.
// 8 waves; wave w computes rows 16w..16w+15. Weights in LDS transposed [N][K].
// A-frag (16x32 f16): lane L holds row (lane&15), K = kb + 8*(L>>4) + {0..7}
//                     and K = kb + 16 + 8*(L>>4) + {0..7}   (two 16B chunks)
// B-frag (32x16 f16): lane L holds col (lane&15), same K layout as A-frag
// C/D   (16x16 f32):  VGPR v <-> row v + 8*(L>>4), col (lane&15)
template<int K_DIM, int N_DIM, int A_STRIDE, int B_STRIDE, int OUT_STRIDE,
         bool RELU, bool OUT_F32>
__device__ __forceinline__ void mlp_layer_tile(
    const _Float16* sA, const _Float16* sW, const float* sBias, void* sOut,
    int lane, int wave)
{
    const int ml = lane & 15;
    const int hi = lane >> 4;

    // Hoist the wave's A fragments: reused across all N_DIM/16 column tiles.
    HFrag a[K_DIM / 32];
    #pragma unroll
    for (int kk = 0; kk < K_DIM / 32; ++kk) {
        const _Float16* pa = sA + (wave * 16 + ml) * A_STRIDE + kk * 32 + hi * 8;
        a[kk].h[0] = *(const v8h*)(pa);
        a[kk].h[1] = *(const v8h*)(pa + 16);
    }

    #pragma unroll
    for (int nt = 0; nt < N_DIM / 16; ++nt) {
        const int n = nt * 16 + ml;
        v8f acc = {0.f, 0.f, 0.f, 0.f, 0.f, 0.f, 0.f, 0.f};
        #pragma unroll
        for (int kk = 0; kk < K_DIM / 32; ++kk) {
            HFrag b;
            const _Float16* pb = sW + n * B_STRIDE + kk * 32 + hi * 8;
            b.h[0] = *(const v8h*)(pb);
            b.h[1] = *(const v8h*)(pb + 16);
            acc = __builtin_amdgcn_wmma_f32_16x16x32_f16(
                false, a[kk].v, false, b.v, (short)0, acc, false, false);
        }
        const float bn = sBias[n];
        const int   mb = wave * 16 + hi * 8;
        #pragma unroll
        for (int v = 0; v < 8; ++v) {
            float x = acc[v] + bn;
            if (RELU) x = fmaxf(x, 0.0f);
            if (OUT_F32)
                ((float*)sOut)[(mb + v) * OUT_STRIDE + n] = x;
            else
                ((_Float16*)sOut)[(mb + v) * OUT_STRIDE + n] = (_Float16)x;
        }
    }
}

// Global f32 weight [K][N] row-major -> LDS f16 transposed [N][BS].
// float4 global reads (coalesced); scalar LDS scatter (transpose).
template<int K, int N, int BS>
__device__ __forceinline__ void load_wt(const float* W, _Float16* sW, int tid) {
    for (int i = tid; i < K * N / 4; i += 256) {
        const int k = (i * 4) / N, n = (i * 4) % N;   // N % 4 == 0, no row cross
        const float4 w = *(const float4*)&W[i * 4];
        sW[(n + 0) * BS + k] = (_Float16)w.x;
        sW[(n + 1) * BS + k] = (_Float16)w.y;
        sW[(n + 2) * BS + k] = (_Float16)w.z;
        sW[(n + 3) * BS + k] = (_Float16)w.w;
    }
}

__device__ __forceinline__ void load_bias(const float* b, float* sB, int n, int tid) {
    for (int i = tid; i < n; i += 256) sB[i] = b[i];
}

// float4 -> 4 packed f16 -> one 8-byte LDS store
__device__ __forceinline__ void store_h4(_Float16* dst, float4 v) {
    HPack4 p;
    p.h[0] = (_Float16)v.x; p.h[1] = (_Float16)v.y;
    p.h[2] = (_Float16)v.z; p.h[3] = (_Float16)v.w;
    *(unsigned long long*)dst = p.u;
}

// ------------------------------- edge kernel -------------------------------
__global__ __launch_bounds__(256) void megnet_edge_kernel(
    const float* __restrict__ bonds, const int* __restrict__ ba1,
    const int* __restrict__ ba2, const float* __restrict__ atoms,
    const float* __restrict__ W1, const float* __restrict__ b1,
    const float* __restrict__ W2, const float* __restrict__ b2,
    const float* __restrict__ W3, const float* __restrict__ b3,
    float* __restrict__ out_bonds, float* __restrict__ seg_sum,
    float* __restrict__ counts, int E)
{
    constexpr int SZ_W1   = 128 * 104 * 2;  // [128][104] f16  (W1^T, K=96)
    constexpr int SZ_W2   = 64  * 136 * 2;  // [64][136]  f16  (W2^T, K=128)
    constexpr int SZ_W3   = 32  * 72  * 2;  // [32][72]   f16  (W3^T, K=64)
    constexpr int SZ_B    = (128 + 64 + 32) * 4;
    constexpr int SZ_IDX  = 128 * 4 * 2;
    constexpr int SZ_BUFA = 128 * 104 * 2;  // Ain [128][104]h, reused: h2 [128][72]h
    constexpr int SZ_BUFB = 128 * 136 * 2;  // h1  [128][136]h, reused: Hf [128][36]f
    __shared__ __align__(16) char smem[SZ_W1 + SZ_W2 + SZ_W3 + SZ_B + SZ_IDX +
                                       SZ_BUFA + SZ_BUFB];
    _Float16* sW1 = (_Float16*)(smem);
    _Float16* sW2 = (_Float16*)(smem + SZ_W1);
    _Float16* sW3 = (_Float16*)(smem + SZ_W1 + SZ_W2);
    float* sB1 = (float*)(smem + SZ_W1 + SZ_W2 + SZ_W3);
    float* sB2 = sB1 + 128;
    float* sB3 = sB2 + 64;
    int* sI1 = (int*)(smem + SZ_W1 + SZ_W2 + SZ_W3 + SZ_B);
    int* sI2 = sI1 + 128;
    char* bufA = smem + SZ_W1 + SZ_W2 + SZ_W3 + SZ_B + SZ_IDX;
    char* bufB = bufA + SZ_BUFA;
    _Float16* sAin = (_Float16*)bufA;   // [128][104]
    _Float16* sH1  = (_Float16*)bufB;   // [128][136]
    _Float16* sH2  = (_Float16*)bufA;   // [128][72]  (aliases Ain, dead by then)
    float*    sHf  = (float*)bufB;      // [128][36]  (aliases H1, dead by then)

    const int tid  = threadIdx.x;
    const int lane = tid & 31, wave = tid >> 5;
    const int e0   = blockIdx.x * 128;

    load_wt<96, 128, 104>(W1, sW1, tid);
    load_wt<128, 64, 136>(W2, sW2, tid);
    load_wt<64, 32, 72>(W3, sW3, tid);
    load_bias(b1, sB1, 128, tid);
    load_bias(b2, sB2, 64, tid);
    load_bias(b3, sB3, 32, tid);
    if (tid < 128) {
        int e = e0 + tid; if (e >= E) e = E - 1;
        sI1[tid] = ba1[e];
        sI2[tid] = ba2[e];
    }
    __syncthreads();

    // Build input tile: [atom1 | atom2 | bond] -> f16 [128][96] (stride 104)
    for (int i = tid; i < 128 * 8; i += 256) {
        const int r = i >> 3, c4 = (i & 7) * 4;
        int e = e0 + r; if (e >= E) e = E - 1;
        const float4 va = *(const float4*)&atoms[sI1[r] * 32 + c4];
        const float4 vb = *(const float4*)&atoms[sI2[r] * 32 + c4];
        const float4 vc = *(const float4*)&bonds[(size_t)e * 32 + c4];
        _Float16* row = sAin + r * 104;
        store_h4(row + c4,      va);
        store_h4(row + 32 + c4, vb);
        store_h4(row + 64 + c4, vc);
    }
    __syncthreads();

    mlp_layer_tile< 96, 128, 104, 104, 136, true,  false>(sAin, sW1, sB1, sH1, lane, wave);
    __syncthreads();
    mlp_layer_tile<128,  64, 136, 136,  72, true,  false>(sH1,  sW2, sB2, sH2, lane, wave);
    __syncthreads();
    mlp_layer_tile< 64,  32,  72,  72,  36, false, true >(sH2,  sW3, sB3, sHf, lane, wave);
    __syncthreads();

    // Write new_bonds + scatter-add onto target atoms (L2-resident f32 atomics)
    for (int i = tid; i < 128 * 8; i += 256) {
        const int r = i >> 3, c4 = (i & 7) * 4;
        const int e = e0 + r;
        if (e < E) {
            const float4 v = *(const float4*)&sHf[r * 36 + c4];
            *(float4*)&out_bonds[(size_t)e * 32 + c4] = v;
            float* s = &seg_sum[sI2[r] * 32 + c4];
            atomicAdd(s + 0, v.x);
            atomicAdd(s + 1, v.y);
            atomicAdd(s + 2, v.z);
            atomicAdd(s + 3, v.w);
        }
    }
    if (tid < 128 && (e0 + tid) < E) atomicAdd(&counts[sI2[tid]], 1.0f);
}

// ------------------------------- node kernel -------------------------------
__global__ __launch_bounds__(256) void megnet_node_kernel(
    const float* __restrict__ atoms,
    const float* __restrict__ W1, const float* __restrict__ b1,
    const float* __restrict__ W2, const float* __restrict__ b2,
    const float* __restrict__ W3, const float* __restrict__ b3,
    const float* __restrict__ seg_sum, const float* __restrict__ counts,
    float* __restrict__ out_atoms, int Nn)
{
    constexpr int SZ_W1   = 128 * 72 * 2;   // [128][72] f16 (W1^T, K=64)
    constexpr int SZ_W2   = 64  * 136 * 2;
    constexpr int SZ_W3   = 32  * 72 * 2;
    constexpr int SZ_B    = (128 + 64 + 32) * 4;
    constexpr int SZ_CNT  = 128 * 4;
    constexpr int SZ_BUFA = 128 * 72 * 2;   // Ain [128][72]h, reused: h2
    constexpr int SZ_BUFB = 128 * 136 * 2;  // h1, reused: Hf [128][36]f
    __shared__ __align__(16) char smem[SZ_W1 + SZ_W2 + SZ_W3 + SZ_B + SZ_CNT +
                                       SZ_BUFA + SZ_BUFB];
    _Float16* sW1 = (_Float16*)(smem);
    _Float16* sW2 = (_Float16*)(smem + SZ_W1);
    _Float16* sW3 = (_Float16*)(smem + SZ_W1 + SZ_W2);
    float* sB1 = (float*)(smem + SZ_W1 + SZ_W2 + SZ_W3);
    float* sB2 = sB1 + 128;
    float* sB3 = sB2 + 64;
    float* sInv = (float*)(smem + SZ_W1 + SZ_W2 + SZ_W3 + SZ_B);
    char* bufA = smem + SZ_W1 + SZ_W2 + SZ_W3 + SZ_B + SZ_CNT;
    char* bufB = bufA + SZ_BUFA;
    _Float16* sAin = (_Float16*)bufA;   // [128][72]
    _Float16* sH1  = (_Float16*)bufB;   // [128][136]
    _Float16* sH2  = (_Float16*)bufA;   // [128][72]
    float*    sHf  = (float*)bufB;      // [128][36]

    const int tid  = threadIdx.x;
    const int lane = tid & 31, wave = tid >> 5;
    const int n0   = blockIdx.x * 128;

    load_wt<64, 128, 72>(W1, sW1, tid);
    load_wt<128, 64, 136>(W2, sW2, tid);
    load_wt<64, 32, 72>(W3, sW3, tid);
    load_bias(b1, sB1, 128, tid);
    load_bias(b2, sB2, 64, tid);
    load_bias(b3, sB3, 32, tid);
    if (tid < 128) {
        int v = n0 + tid; if (v >= Nn) v = Nn - 1;
        sInv[tid] = 1.0f / fmaxf(counts[v], 1.0f);
    }
    __syncthreads();

    // Build input tile: [mean_bonds | atoms] -> f16 [128][64] (stride 72)
    for (int i = tid; i < 128 * 8; i += 256) {
        const int r = i >> 3, c4 = (i & 7) * 4;
        int v = n0 + r; if (v >= Nn) v = Nn - 1;
        float4 vs = *(const float4*)&seg_sum[v * 32 + c4];
        const float inv = sInv[r];
        vs.x *= inv; vs.y *= inv; vs.z *= inv; vs.w *= inv;
        const float4 va = *(const float4*)&atoms[v * 32 + c4];
        _Float16* row = sAin + r * 72;
        store_h4(row + c4,      vs);
        store_h4(row + 32 + c4, va);
    }
    __syncthreads();

    mlp_layer_tile< 64, 128,  72,  72, 136, true,  false>(sAin, sW1, sB1, sH1, lane, wave);
    __syncthreads();
    mlp_layer_tile<128,  64, 136, 136,  72, true,  false>(sH1,  sW2, sB2, sH2, lane, wave);
    __syncthreads();
    mlp_layer_tile< 64,  32,  72,  72,  36, false, true >(sH2,  sW3, sB3, sHf, lane, wave);
    __syncthreads();

    for (int i = tid; i < 128 * 8; i += 256) {
        const int r = i >> 3, c4 = (i & 7) * 4;
        const int v = n0 + r;
        if (v < Nn)
            *(float4*)&out_atoms[(size_t)v * 32 + c4] = *(const float4*)&sHf[r * 36 + c4];
    }
}

// ------------------------------- launcher ----------------------------------
extern "C" void kernel_launch(void* const* d_in, const int* in_sizes, int n_in,
                              void* d_out, int out_size, void* d_ws, size_t ws_size,
                              hipStream_t stream) {
    const float* bonds = (const float*)d_in[0];
    const int*   ba1   = (const int*)d_in[1];
    const int*   ba2   = (const int*)d_in[2];
    const float* atoms = (const float*)d_in[3];
    const float* We1 = (const float*)d_in[4];
    const float* be1 = (const float*)d_in[5];
    const float* We2 = (const float*)d_in[6];
    const float* be2 = (const float*)d_in[7];
    const float* We3 = (const float*)d_in[8];
    const float* be3 = (const float*)d_in[9];
    const float* Wv1 = (const float*)d_in[10];
    const float* bv1 = (const float*)d_in[11];
    const float* Wv2 = (const float*)d_in[12];
    const float* bv2 = (const float*)d_in[13];
    const float* Wv3 = (const float*)d_in[14];
    const float* bv3 = (const float*)d_in[15];

    const int E  = in_sizes[1];        // number of edges
    const int Nn = in_sizes[3] / 32;   // number of nodes

    float* out_atoms = (float*)d_out;                       // [Nn, 32]
    float* out_bonds = out_atoms + (size_t)Nn * 32;         // [E, 32]
    float* seg = (float*)d_ws;                              // [Nn, 32]
    float* cnt = seg + (size_t)Nn * 32;                     // [Nn]

    hipMemsetAsync(d_ws, 0, ((size_t)Nn * 32 + Nn) * sizeof(float), stream);

    megnet_edge_kernel<<<(E + 127) / 128, 256, 0, stream>>>(
        bonds, ba1, ba2, atoms, We1, be1, We2, be2, We3, be3,
        out_bonds, seg, cnt, E);

    megnet_node_kernel<<<(Nn + 127) / 128, 256, 0, stream>>>(
        atoms, Wv1, bv1, Wv2, bv2, Wv3, bv3, seg, cnt, out_atoms, Nn);
}
